// AttentionLayer_57303453663823
// MI455X (gfx1250) — matrix operand order repaired and checked by reading, not compiled, not run on hardware
//
#include <hip/hip_runtime.h>

#define HID 1024
#define NH  16
#define HD  64
#define SEQ 2048
#define NB  2
#define ROTD 32

typedef __bf16 bf16_t;
typedef __attribute__((ext_vector_type(16))) bf16_t v16bf;
typedef __attribute__((ext_vector_type(8)))  float  v8f;
typedef int v4i_t __attribute__((vector_size(16)));

union Frag32 { uint4 u4[2]; v16bf v; };

#define GAS __attribute__((address_space(1)))
#define LAS __attribute__((address_space(3)))

#if defined(__gfx1250__) && __has_builtin(__builtin_amdgcn_global_load_async_to_lds_b128)
#define USE_ASYNC_LDS 1
#else
#define USE_ASYNC_LDS 0
#endif

__device__ __forceinline__ void wait_async_lds() {
#if USE_ASYNC_LDS
#if __has_builtin(__builtin_amdgcn_s_wait_asynccnt)
    __builtin_amdgcn_s_wait_asynccnt(0);
#else
    asm volatile("s_wait_asynccnt 0x0" ::: "memory");
#endif
#endif
}

#if USE_ASYNC_LDS
__device__ __forceinline__ void async_copy16(const unsigned short* g,
                                             unsigned short* l) {
    __builtin_amdgcn_global_load_async_to_lds_b128(
        (GAS v4i_t*)g, (LAS v4i_t*)l, 0, 0);
}
#endif

__device__ __forceinline__ unsigned short f2bf(float f) {
    unsigned int u = __float_as_uint(f);
    u += 0x7FFFu + ((u >> 16) & 1u);   // round-to-nearest-even
    return (unsigned short)(u >> 16);
}

__device__ __forceinline__ v8f zero8() {
    v8f z = {0.f,0.f,0.f,0.f,0.f,0.f,0.f,0.f};
    return z;
}

// ---------------------------------------------------------------------------
// fp32 -> bf16 elementwise convert (for x)
// ---------------------------------------------------------------------------
__global__ void cvt_f32_bf16(const float* __restrict__ in,
                             unsigned short* __restrict__ out, int n) {
    int i = blockIdx.x * blockDim.x + threadIdx.x;
    if (i < n) out[i] = f2bf(in[i]);
}

// ---------------------------------------------------------------------------
// fp32 -> bf16 convert + transpose (for HIDxHID weights): out[n][k] = in[k][n].
// LDS-tiled so both global reads and writes are coalesced.
// ---------------------------------------------------------------------------
__global__ __launch_bounds__(256)
void cvt_tr_f32_bf16(const float* __restrict__ in,
                     unsigned short* __restrict__ out) {
    __shared__ float tile[32][33];
    const int x0 = blockIdx.x * 32;   // input col (n)
    const int y0 = blockIdx.y * 32;   // input row (k)
    const int tx = threadIdx.x;       // 0..31
    const int ty = threadIdx.y;       // 0..7
#pragma unroll
    for (int j = 0; j < 32; j += 8)
        tile[ty + j][tx] = in[(size_t)(y0 + ty + j) * HID + x0 + tx];
    __syncthreads();
#pragma unroll
    for (int j = 0; j < 32; j += 8)
        out[(size_t)(x0 + ty + j) * HID + y0 + tx] = f2bf(tile[tx][ty + j]);
}

// ---------------------------------------------------------------------------
// C = A(MxK) @ W(KxN) + bias, where Bt is W pre-transposed to NxK (bf16).
// 256 threads = 8 waves; block tile 128x128; each wave owns a 16x128 strip.
// Both tiles stage via async-to-LDS (plain row-major copies, no transform),
// double-buffered; fragments are b128 loads on padded (40-half) strides; the
// 8 WMMAs per k-step chain on independent accumulators.
// ---------------------------------------------------------------------------
__global__ __launch_bounds__(256)
void gemm_bf16(const unsigned short* __restrict__ A,
               const unsigned short* __restrict__ Bt,
               const float* __restrict__ bias,
               float* __restrict__ C,
               int M, int N, int K)
{
    __shared__ __align__(16) unsigned short sA[2][128 * 40]; // [m][k]
    __shared__ __align__(16) unsigned short sB[2][128 * 40]; // [n][k]

    const int t     = threadIdx.x;
    const int lane  = t & 31;
    const int wave  = t >> 5;
    const int m0    = blockIdx.y * 128;
    const int n0    = blockIdx.x * 128;
    const int mlane = lane & 15;
    const int kbq   = (lane >> 4) << 4;   // A-frag half base: 0 / 16
    const int koff  = (lane >> 4) << 4;   // B-frag half base: 0 / 16

    // cooperative staging coordinates (identical for both tiles)
    const int sr = t >> 1;                // 0..127 (tile row)
    const int sc = (t & 1) << 4;          // 0 / 16 (k-chunk)

    const unsigned short* gA = A  + (size_t)(m0 + sr) * K + sc;
    const unsigned short* gB = Bt + (size_t)(n0 + sr) * K + sc;

    v8f acc[8];
#pragma unroll
    for (int i = 0; i < 8; i++) acc[i] = zero8();

    // ---- prologue: stage tile 0 into buffer 0 ----
    {
#if USE_ASYNC_LDS
        async_copy16(gA,     &sA[0][sr * 40 + sc]);
        async_copy16(gA + 8, &sA[0][sr * 40 + sc + 8]);
        async_copy16(gB,     &sB[0][sr * 40 + sc]);
        async_copy16(gB + 8, &sB[0][sr * 40 + sc + 8]);
        wait_async_lds();
#else
        uint4 a0 = *(const uint4*)gA;
        uint4 a1 = *(const uint4*)(gA + 8);
        uint4 b0 = *(const uint4*)gB;
        uint4 b1 = *(const uint4*)(gB + 8);
        *(uint4*)&sA[0][sr * 40 + sc]     = a0;
        *(uint4*)&sA[0][sr * 40 + sc + 8] = a1;
        *(uint4*)&sB[0][sr * 40 + sc]     = b0;
        *(uint4*)&sB[0][sr * 40 + sc + 8] = b1;
#endif
    }
    __syncthreads();

    const int KSTEPS = K >> 5;
    for (int ks = 0; ks < KSTEPS; ks++) {
        const int cur = ks & 1;
        const int nxt = cur ^ 1;
        const bool hasnext = (ks + 1) < KSTEPS;

        // ---- issue prefetch for tile ks+1 (latency overlapped) ----
#if !USE_ASYNC_LDS
        uint4 na0, na1, nb0, nb1;
#endif
        if (hasnext) {
            const unsigned short* gA2 = gA + (size_t)(ks + 1) * 32;
            const unsigned short* gB2 = gB + (size_t)(ks + 1) * 32;
#if USE_ASYNC_LDS
            async_copy16(gA2,     &sA[nxt][sr * 40 + sc]);
            async_copy16(gA2 + 8, &sA[nxt][sr * 40 + sc + 8]);
            async_copy16(gB2,     &sB[nxt][sr * 40 + sc]);
            async_copy16(gB2 + 8, &sB[nxt][sr * 40 + sc + 8]);
#else
            na0 = *(const uint4*)gA2;
            na1 = *(const uint4*)(gA2 + 8);
            nb0 = *(const uint4*)gB2;
            nb1 = *(const uint4*)(gB2 + 8);
#endif
        }

        // ---- compute on buffer cur: preload frags, then WMMA chain ----
        Frag32 af;
        af.u4[0] = *(const uint4*)&sA[cur][(wave * 16 + mlane) * 40 + kbq];
        af.u4[1] = *(const uint4*)&sA[cur][(wave * 16 + mlane) * 40 + kbq + 8];
        Frag32 bf8[8];
#pragma unroll
        for (int nt = 0; nt < 8; nt++) {
            const int n = nt * 16 + mlane;
            bf8[nt].u4[0] = *(const uint4*)&sB[cur][n * 40 + koff];
            bf8[nt].u4[1] = *(const uint4*)&sB[cur][n * 40 + koff + 8];
        }
#pragma unroll
        for (int nt = 0; nt < 8; nt++)
            acc[nt] = __builtin_amdgcn_wmma_f32_16x16x32_bf16(
                false, af.v, false, bf8[nt].v, (short)0, acc[nt], false, false);

        // ---- land prefetched data into buffer nxt ----
        if (hasnext) {
#if USE_ASYNC_LDS
            wait_async_lds();
#else
            *(uint4*)&sA[nxt][sr * 40 + sc]     = na0;
            *(uint4*)&sA[nxt][sr * 40 + sc + 8] = na1;
            *(uint4*)&sB[nxt][sr * 40 + sc]     = nb0;
            *(uint4*)&sB[nxt][sr * 40 + sc + 8] = nb1;
#endif
        }
        __syncthreads();
    }

    // ---- epilogue: C layout VGPR r -> row r (+8 for upper lanes) ----
    const int rbase = (lane >> 4) << 3;
#pragma unroll
    for (int nt = 0; nt < 8; nt++) {
        const int col = n0 + nt * 16 + mlane;
        const float bv = bias[col];
#pragma unroll
        for (int r = 0; r < 8; r++) {
            const int row = m0 + wave * 16 + r + rbase;
            C[(size_t)row * N + col] = acc[nt][r] + bv;
        }
    }
}

// ---------------------------------------------------------------------------
// Rotary (element-wise per the reference), q scale, bf16 pack.
// T=0 -> Qb row-major, T=1 -> Kb row-major, T=2 -> V transposed (b,h,d,s).
// ---------------------------------------------------------------------------
__global__ void rotary_pack(const float* __restrict__ qkvf,
                            const float* __restrict__ sinus,
                            unsigned short* __restrict__ qb,
                            unsigned short* __restrict__ kb,
                            unsigned short* __restrict__ vt)
{
    const int TSZ = NB * SEQ * HID;
    int i = blockIdx.x * blockDim.x + threadIdx.x;
    int T = i / TSZ;
    int e = i - T * TSZ;
    int row = e >> 10;          // b*SEQ + s
    int col = e & 1023;
    int b = row >> 11;
    int s = row & 2047;
    int d = col & 63;
    float v = qkvf[i];
    if (T < 2 && d < ROTD) {
        float sn = sinus[((b * 2 + 0) * SEQ + s) * ROTD + d];
        float cs = sinus[((b * 2 + 1) * SEQ + s) * ROTD + d];
        float sign = (d & 1) ? 1.0f : -1.0f;
        v = v * cs + sign * v * sn;
    }
    if (T == 0) {
        v *= 0.125f;            // 1/sqrt(D)
        qb[e] = f2bf(v);
    } else if (T == 1) {
        kb[e] = f2bf(v);
    } else {
        int h = col >> 6;
        vt[(((size_t)b * NH + h) * HD + d) * SEQ + s] = f2bf(v);
    }
}

// ---------------------------------------------------------------------------
// Flash attention: 1 wave = 16 query rows of one (b,h); 32 keys / iteration.
// Scores via Q·K^T WMMA (K/V are L2-resident), online softmax with shfl_xor
// row-reductions, P relayout through per-wave LDS, O += P·V WMMA.
// ---------------------------------------------------------------------------
__global__ __launch_bounds__(128)
void flash_attn(const unsigned short* __restrict__ Qb,
                const unsigned short* __restrict__ Kb,
                const unsigned short* __restrict__ Vt,
                const float* __restrict__ abias,
                unsigned short* __restrict__ Ob)
{
    __shared__ __align__(16) unsigned short sP[4][16 * 40];

    const int lane  = threadIdx.x & 31;
    const int wslot = threadIdx.x >> 5;
    const int w     = blockIdx.x * 4 + wslot;
    const int qt    = w & 127;
    const int h     = (w >> 7) & 15;
    const int b     = w >> 11;
    const int q0    = qt * 16;
    const int mlane = lane & 15;
    const int kbq   = (lane >> 4) << 4;   // A-frag half base
    const int khalf = (lane >> 4) << 4;   // B-frag half base
    const int rbase = (lane >> 4) << 3;

    // Q A-fragments for both 32-wide D chunks (kept in registers)
    Frag32 qa[2];
    {
        const unsigned short* qrow =
            Qb + (size_t)(b * SEQ + q0 + mlane) * HID + h * HD;
#pragma unroll
        for (int c = 0; c < 2; c++) {
            qa[c].u4[0] = *(const uint4*)(qrow + c * 32 + kbq);
            qa[c].u4[1] = *(const uint4*)(qrow + c * 32 + kbq + 8);
        }
    }

    v8f o[4];
#pragma unroll
    for (int i = 0; i < 4; i++) o[i] = zero8();
    float mr[8], lr[8];
#pragma unroll
    for (int r = 0; r < 8; r++) { mr[r] = -1e30f; lr[r] = 0.0f; }

    unsigned short* P = sP[wslot];

    for (int kt = 0; kt < SEQ / 32; kt++) {
        const int krow0 = kt * 32;

        // ---- scores: two 16x16 tiles over this key block ----
        Frag32 kf[2][2];
#pragma unroll
        for (int jt = 0; jt < 2; jt++) {
            const unsigned short* krow =
                Kb + (size_t)(b * SEQ + krow0 + jt * 16 + mlane) * HID + h * HD;
#pragma unroll
            for (int ch = 0; ch < 2; ch++) {
                kf[jt][ch].u4[0] = *(const uint4*)(krow + ch * 32 + khalf);
                kf[jt][ch].u4[1] = *(const uint4*)(krow + ch * 32 + khalf + 8);
            }
        }
        v8f s[2];
#pragma unroll
        for (int jt = 0; jt < 2; jt++) {
            v8f c = zero8();
#pragma unroll
            for (int ch = 0; ch < 2; ch++)
                c = __builtin_amdgcn_wmma_f32_16x16x32_bf16(
                    false, qa[ch].v, false, kf[jt][ch].v, (short)0, c,
                    false, false);
            s[jt] = c;
        }

        // ---- add attention bias (B,1,S,S) ----
#pragma unroll
        for (int jt = 0; jt < 2; jt++) {
            const float* bp = abias + ((size_t)b * SEQ + q0) * SEQ
                            + krow0 + jt * 16 + mlane;
#pragma unroll
            for (int r = 0; r < 8; r++)
                s[jt][r] += bp[(size_t)(r + rbase) * SEQ];
        }

        // ---- online softmax over 32 key columns ----
        float alpha[8];
#pragma unroll
        for (int r = 0; r < 8; r++) {
            float tmax = fmaxf(s[0][r], s[1][r]);
            tmax = fmaxf(tmax, __shfl_xor(tmax, 1, 32));
            tmax = fmaxf(tmax, __shfl_xor(tmax, 2, 32));
            tmax = fmaxf(tmax, __shfl_xor(tmax, 4, 32));
            tmax = fmaxf(tmax, __shfl_xor(tmax, 8, 32));
            float mnew = fmaxf(mr[r], tmax);
            alpha[r] = __expf(mr[r] - mnew);
            mr[r] = mnew;
            float p0 = __expf(s[0][r] - mnew);
            float p1 = __expf(s[1][r] - mnew);
            s[0][r] = p0; s[1][r] = p1;
            float ps = p0 + p1;
            ps += __shfl_xor(ps, 1, 32);
            ps += __shfl_xor(ps, 2, 32);
            ps += __shfl_xor(ps, 4, 32);
            ps += __shfl_xor(ps, 8, 32);
            lr[r] = lr[r] * alpha[r] + ps;
        }

        // ---- rescale running O ----
#pragma unroll
        for (int nt = 0; nt < 4; nt++)
#pragma unroll
            for (int r = 0; r < 8; r++) o[nt][r] *= alpha[r];

        // ---- P: C-layout -> LDS (bf16) -> A-layout (same-wave, in-order) ----
#pragma unroll
        for (int jt = 0; jt < 2; jt++)
#pragma unroll
            for (int r = 0; r < 8; r++)
                P[(r + rbase) * 40 + jt * 16 + mlane] = f2bf(s[jt][r]);
        asm volatile("" ::: "memory");

        Frag32 pf;
        pf.u4[0] = *(const uint4*)(&P[mlane * 40 + kbq]);
        pf.u4[1] = *(const uint4*)(&P[mlane * 40 + kbq + 8]);

        // ---- O += P·V  (V transposed layout: contiguous key runs) ----
        const unsigned short* vbase =
            Vt + (size_t)(b * NH + h) * HD * SEQ + krow0 + khalf;
        Frag32 vf[4];
#pragma unroll
        for (int nt = 0; nt < 4; nt++) {
            const unsigned short* vp = vbase + (size_t)(nt * 16 + mlane) * SEQ;
            vf[nt].u4[0] = *(const uint4*)(vp);
            vf[nt].u4[1] = *(const uint4*)(vp + 8);
        }
#pragma unroll
        for (int nt = 0; nt < 4; nt++)
            o[nt] = __builtin_amdgcn_wmma_f32_16x16x32_bf16(
                false, pf.v, false, vf[nt].v, (short)0, o[nt], false, false);
        asm volatile("" ::: "memory");
    }

    // ---- normalize and store bf16 (row-major (b,s,h,d)) ----
#pragma unroll
    for (int nt = 0; nt < 4; nt++)
#pragma unroll
        for (int r = 0; r < 8; r++) {
            const int row = q0 + r + rbase;
            const float val = o[nt][r] / lr[r];
            Ob[(size_t)(b * SEQ + row) * HID + h * HD + nt * 16 + mlane] =
                f2bf(val);
        }
}

// ---------------------------------------------------------------------------
extern "C" void kernel_launch(void* const* d_in, const int* in_sizes, int n_in,
                              void* d_out, int out_size, void* d_ws,
                              size_t ws_size, hipStream_t stream) {
    (void)in_sizes; (void)n_in; (void)out_size; (void)ws_size;
    const float* x     = (const float*)d_in[0];
    const float* sinus = (const float*)d_in[1];
    const float* abias = (const float*)d_in[2];
    const float* Wq    = (const float*)d_in[3];
    const float* bq    = (const float*)d_in[4];
    const float* Wk    = (const float*)d_in[5];
    const float* bk    = (const float*)d_in[6];
    const float* Wv    = (const float*)d_in[7];
    const float* bv    = (const float*)d_in[8];
    const float* Wo    = (const float*)d_in[9];
    float* out = (float*)d_out;

    char* ws = (char*)d_ws;
    const size_t TSZ = (size_t)NB * SEQ * HID;   // 4,194,304
    const size_t WSZ = (size_t)HID * HID;        // 1,048,576
    unsigned short* xb   = (unsigned short*)ws;                        // 8 MB
    unsigned short* wb   = (unsigned short*)(ws + (8ull << 20));       // 8 MB (transposed)
    float*          qkvf = (float*)(ws + (16ull << 20));               // 48 MB
    unsigned short* qb   = (unsigned short*)(ws + (64ull << 20));      // 24 MB
    unsigned short* kb2  = qb + TSZ;
    unsigned short* vt   = qb + 2 * TSZ;
    float*          zb   = (float*)(ws + (88ull << 20));               // 4 KB
    unsigned short* attnb = xb;   // reuse: x no longer needed post-QKV

    (void)hipMemsetAsync(zb, 0, HID * sizeof(float), stream);

    // 1) fp32 -> bf16 (x plain; weights transposed to NxK for async staging)
    cvt_f32_bf16<<<(int)(TSZ / 256), 256, 0, stream>>>(x, xb, (int)TSZ);
    dim3 tb(32, 8), tg(HID / 32, HID / 32);
    cvt_tr_f32_bf16<<<tg, tb, 0, stream>>>(Wq, wb + 0 * WSZ);
    cvt_tr_f32_bf16<<<tg, tb, 0, stream>>>(Wk, wb + 1 * WSZ);
    cvt_tr_f32_bf16<<<tg, tb, 0, stream>>>(Wv, wb + 2 * WSZ);
    cvt_tr_f32_bf16<<<tg, tb, 0, stream>>>(Wo, wb + 3 * WSZ);

    // 2) Q/K/V projections
    dim3 gg(HID / 128, (NB * SEQ) / 128);   // (8, 32)
    gemm_bf16<<<gg, 256, 0, stream>>>(xb, wb + 0 * WSZ, bq, qkvf + 0 * TSZ,
                                      NB * SEQ, HID, HID);
    gemm_bf16<<<gg, 256, 0, stream>>>(xb, wb + 1 * WSZ, bk, qkvf + 1 * TSZ,
                                      NB * SEQ, HID, HID);
    gemm_bf16<<<gg, 256, 0, stream>>>(xb, wb + 2 * WSZ, bv, qkvf + 2 * TSZ,
                                      NB * SEQ, HID, HID);

    // 3) rotary + scale + bf16 pack (V transposed)
    rotary_pack<<<(int)(3 * TSZ / 256), 256, 0, stream>>>(qkvf, sinus,
                                                          qb, kb2, vt);

    // 4) flash attention
    flash_attn<<<(NB * NH * (SEQ / 16)) / 4, 128, 0, stream>>>(qb, kb2, vt,
                                                               abias, attnb);

    // 5) output projection (zero bias vector keeps epilogue branch-free)
    gemm_bf16<<<gg, 256, 0, stream>>>(attnb, wb + 3 * WSZ, zb, out,
                                      NB * SEQ, HID, HID);
}
